// ClassifierAttnMulti_8701603742240
// MI455X (gfx1250) — compile-verified, matrix operand order checked by reading, not run on hardware
//
#include <hip/hip_runtime.h>

#define N_NODES   100000
#define N_EDGES   1600000
#define N_GRAPHS  64
#define IN_DIM    128
#define HID       64
#define N_CLASSES 10

typedef __attribute__((ext_vector_type(16))) __bf16 v16bf;
typedef __attribute__((ext_vector_type(8)))  float  v8f;

union Frag16 {
    v16bf v;
    unsigned short us[16];
    unsigned u[8];
};

__device__ __forceinline__ unsigned short f2bf(float f) {
    unsigned u = __float_as_uint(f);
    unsigned r = u + 0x7FFFu + ((u >> 16) & 1u);   // round-to-nearest-even
    return (unsigned short)(r >> 16);
}
__device__ __forceinline__ float bf2f(unsigned short b) {
    return __uint_as_float(((unsigned)b) << 16);
}

// ---------------------------------------------------------------- utilities
__global__ void zero_f32(float* __restrict__ p, int n) {
    int i = blockIdx.x * blockDim.x + threadIdx.x;
    if (i < n) p[i] = 0.0f;
}

__global__ void f32_to_bf16(const float* __restrict__ src,
                            unsigned short* __restrict__ dst, int n) {
    int i = blockIdx.x * blockDim.x + threadIdx.x;
    if (i < n) dst[i] = f2bf(src[i]);
}

// -------------------------------------------------- W pre-swizzle (cast+pack)
// Packs W[K,64] (f32, row-major) into the exact B-fragment dword layout:
//   out[((kt*4 + wave)*32 + lane)*8 + p]  =  {bf16(W[K0,n]), bf16(W[K0+1,n])}
// with half = lane>>4, l15 = lane&15, kb = 8*half,
//      K0 = kt*32 + kb + 2p + (p>=4 ? 8 : 0),  n = wave*16 + l15.
// The GEMM then loads B as 8 contiguous dwords per lane (2x b128).
__global__ void swizzle_W(const float* __restrict__ W, int K,
                          unsigned* __restrict__ out) {
    int idx = blockIdx.x * blockDim.x + threadIdx.x;
    int total = (K / 32) * 4 * 32 * 8;
    if (idx >= total) return;
    int p    = idx & 7;
    int lane = (idx >> 3) & 31;
    int wv   = (idx >> 8) & 3;
    int kt   = idx >> 10;
    int half = lane >> 4;
    int l15  = lane & 15;
    int kb   = half * 8;
    int K0   = kt * 32 + kb + 2 * p + ((p >= 4) ? 8 : 0);
    int n    = wv * 16 + l15;
    unsigned lo = f2bf(W[(size_t)K0 * HID + n]);
    unsigned hi = f2bf(W[(size_t)(K0 + 1) * HID + n]);
    out[idx] = lo | (hi << 16);
}

// ------------------------------------------------------------------ degrees
__global__ void degree_kernel(const int* __restrict__ ei,
                              float* __restrict__ deg_out,
                              float* __restrict__ deg_in) {
    int e = blockIdx.x * blockDim.x + threadIdx.x;
    if (e >= N_EDGES) return;
    atomicAdd(deg_out + ei[e], 1.0f);            // src
    atomicAdd(deg_in + ei[N_EDGES + e], 1.0f);   // dst
}

__global__ void rsqrt_clip_kernel(float* __restrict__ p, int n) {
    int i = blockIdx.x * blockDim.x + threadIdx.x;
    if (i < n) {
        float d = p[i];
        d = fmaxf(d, 1.0f);
        p[i] = rsqrtf(d);
    }
}

// ---------------------------------------------------------------- WMMA GEMM
// out[node, 0:64] = bf16( (A[node, 0:K] @ W[0:K, 0:64]) * rowScale[node] )
// Block = 128 threads (4 waves); wave w owns the 16x16 C tile at N = 16*w.
// Grid.x = N_NODES/16 = 6250.  K templated -> fully unrolled (K/32 WMMAs).
template <int K>
__global__ void gemm_nodes_bf16(const unsigned short* __restrict__ A,
                                const unsigned* __restrict__ Wsw,
                                const float* __restrict__ rowScale,
                                unsigned short* __restrict__ out) {
    const int wave = threadIdx.x >> 5;
    const int lane = threadIdx.x & 31;
    const int half = lane >> 4;     // 0: K base 0, 1: K base +8 (ISA layout)
    const int l15  = lane & 15;
    const int kb   = half * 8;
    const int m0   = blockIdx.x * 16;
    const int nb   = wave * 16;

    v8f c = {0.f, 0.f, 0.f, 0.f, 0.f, 0.f, 0.f, 0.f};
    const unsigned short* arow = A + (size_t)(m0 + l15) * K;

    #pragma unroll
    for (int kt = 0; kt < K / 32; ++kt) {
        Frag16 a, b;
        // A fragment: pairs p<4 and p>=4 each cover 16 contiguous bytes.
        #pragma unroll
        for (int p = 0; p < 8; ++p) {
            int koff = kt * 32 + kb + 2 * p + ((p >= 4) ? 8 : 0);
            a.u[p] = *(const unsigned*)(arow + koff);
        }
        // B fragment: pre-swizzled, 8 contiguous dwords per lane.
        const unsigned* bptr = Wsw + ((size_t)(kt * 4 + wave) * 32 + lane) * 8;
        #pragma unroll
        for (int p = 0; p < 8; ++p) b.u[p] = bptr[p];

        c = __builtin_amdgcn_wmma_f32_16x16x32_bf16(
                false, a.v, false, b.v, (short)0, c, false, false);
    }

    // C tile: VGPR r holds row m0 + r + 8*half, column nb + l15.
    #pragma unroll
    for (int r = 0; r < 8; ++r) {
        int node = m0 + r + 8 * half;
        float v = c[r] * rowScale[node];
        out[(size_t)node * HID + nb + l15] = f2bf(v);
    }
}

// --------------------------------------------------------------------- SpMM
// m[dst, :] += t[src, :]   (messages pre-scaled by norm_out)
// One wave per edge, 2 channels per lane (packed bf16 load, f32 atomics).
__global__ void spmm_kernel(const int* __restrict__ ei,
                            const unsigned short* __restrict__ t,
                            float* __restrict__ m) {
    long long tid = (long long)blockIdx.x * blockDim.x + threadIdx.x;
    if (tid >= (long long)N_EDGES * 32) return;
    int e  = (int)(tid >> 5);
    int c2 = ((int)tid & 31) * 2;
    int src = ei[e];
    int dst = ei[N_EDGES + e];
    unsigned pk = *(const unsigned*)(t + (size_t)src * HID + c2);
    float f0 = bf2f((unsigned short)(pk & 0xFFFFu));
    float f1 = bf2f((unsigned short)(pk >> 16));
    float* d = m + (size_t)dst * HID + c2;
    atomicAdd(d,     f0);
    atomicAdd(d + 1, f1);
}

// ------------------------------------------------ layer epilogue: relu(...)
__global__ void finish_layer_kernel(const float* __restrict__ m,
                                    const float* __restrict__ norm_in,
                                    const float* __restrict__ bias,
                                    unsigned short* __restrict__ h) {
    int i = blockIdx.x * blockDim.x + threadIdx.x;
    if (i >= N_NODES * HID) return;
    int node = i >> 6;   // /HID
    int c    = i & 63;
    float v = m[i] * norm_in[node] + bias[c];
    h[i] = f2bf(fmaxf(v, 0.0f));
}

// --------------------- layer-2 epilogue + attention gate + per-graph pooling
// One wave per node, 2 channels per lane; wave32 shuffle reduction for the
// attention dot product.
__global__ void finish2_attn_pool_kernel(const float* __restrict__ m2,
                                         const float* __restrict__ norm_in,
                                         const float* __restrict__ b2,
                                         const float* __restrict__ attn_W,
                                         const float* __restrict__ attn_b,
                                         const int* __restrict__ graph_id,
                                         float* __restrict__ hw_out,
                                         float* __restrict__ gsum,
                                         float* __restrict__ gcnt) {
    int wave = threadIdx.x >> 5;
    int lane = threadIdx.x & 31;
    int node = blockIdx.x * 8 + wave;
    if (node >= N_NODES) return;
    int c0 = lane * 2;
    const float ni = norm_in[node];
    float v0 = fmaxf(m2[(size_t)node * HID + c0]     * ni + b2[c0],     0.0f);
    float v1 = fmaxf(m2[(size_t)node * HID + c0 + 1] * ni + b2[c0 + 1], 0.0f);

    float dot = v0 * attn_W[c0] + v1 * attn_W[c0 + 1];
    #pragma unroll
    for (int o = 16; o > 0; o >>= 1) dot += __shfl_xor(dot, o, 32);
    float gate = 1.0f / (1.0f + __expf(-(dot + attn_b[0])));

    if (lane == 0) hw_out[node] = gate;

    int g = graph_id[node];
    atomicAdd(gsum + (size_t)g * HID + c0,     v0 * gate);
    atomicAdd(gsum + (size_t)g * HID + c0 + 1, v1 * gate);
    if (lane == 0) atomicAdd(gcnt + g, 1.0f);
}

// ----------------------------------------------------------- tiny classifier
__global__ void classify_kernel(const float* __restrict__ gsum,
                                const float* __restrict__ gcnt,
                                const float* __restrict__ cls_W,
                                const float* __restrict__ cls_b,
                                float* __restrict__ out) {
    int tid = threadIdx.x;
    if (tid >= N_GRAPHS * N_CLASSES) return;
    int g = tid / N_CLASSES;
    int c = tid % N_CLASSES;
    float inv = 1.0f / fmaxf(gcnt[g], 1.0f);
    float acc = cls_b[c];
    #pragma unroll
    for (int k = 0; k < HID; ++k)
        acc += (gsum[g * HID + k] * inv) * cls_W[k * N_CLASSES + c];
    out[g * N_CLASSES + c] = acc;
}

// =============================================================== launcher
extern "C" void kernel_launch(void* const* d_in, const int* in_sizes, int n_in,
                              void* d_out, int out_size, void* d_ws, size_t ws_size,
                              hipStream_t stream) {
    const float* x       = (const float*)d_in[0];
    const int*   ei      = (const int*)d_in[1];
    const int*   gid     = (const int*)d_in[2];
    const float* W1      = (const float*)d_in[3];
    const float* b1      = (const float*)d_in[4];
    const float* W2      = (const float*)d_in[5];
    const float* b2      = (const float*)d_in[6];
    const float* attn_W  = (const float*)d_in[7];
    const float* attn_b  = (const float*)d_in[8];
    const float* cls_W   = (const float*)d_in[9];
    const float* cls_b   = (const float*)d_in[10];

    float* out_cls = (float*)d_out;                         // [64, 10]
    float* out_hw  = (float*)d_out + N_GRAPHS * N_CLASSES;  // [100000]

    // ---- workspace carve-up (256B aligned slices)
    char* w = (char*)d_ws;
    auto alloc = [&](size_t bytes) {
        char* p = w;
        w += (bytes + 255) & ~(size_t)255;
        return p;
    };
    float*          norm_out = (float*)alloc((size_t)N_NODES * 4);
    float*          norm_in  = (float*)alloc((size_t)N_NODES * 4);
    unsigned short* xb       = (unsigned short*)alloc((size_t)N_NODES * IN_DIM * 2);
    unsigned*       w1s      = (unsigned*)alloc((size_t)(IN_DIM / 32) * 1024 * 4); // swizzled B frags
    unsigned*       w2s      = (unsigned*)alloc((size_t)(HID / 32) * 1024 * 4);
    unsigned short* t        = (unsigned short*)alloc((size_t)N_NODES * HID * 2);  // messages (bf16)
    float*          m        = (float*)alloc((size_t)N_NODES * HID * 4);           // aggregation (f32)
    unsigned short* h1       = (unsigned short*)alloc((size_t)N_NODES * HID * 2);
    float*          gsum     = (float*)alloc((size_t)N_GRAPHS * HID * 4);
    float*          gcnt     = (float*)alloc((size_t)N_GRAPHS * 4);

    const int B = 256;
    auto nb = [](long long n, int b) { return (unsigned)((n + b - 1) / b); };

    // ---- degrees -> rsqrt norms
    zero_f32<<<nb(N_NODES, B), B, 0, stream>>>(norm_out, N_NODES);
    zero_f32<<<nb(N_NODES, B), B, 0, stream>>>(norm_in, N_NODES);
    degree_kernel<<<nb(N_EDGES, B), B, 0, stream>>>(ei, norm_out, norm_in);
    rsqrt_clip_kernel<<<nb(N_NODES, B), B, 0, stream>>>(norm_out, N_NODES);
    rsqrt_clip_kernel<<<nb(N_NODES, B), B, 0, stream>>>(norm_in, N_NODES);

    // ---- bf16 cast of x, swizzled cast of weights
    f32_to_bf16<<<nb((long long)N_NODES * IN_DIM, B), B, 0, stream>>>(x, xb, N_NODES * IN_DIM);
    swizzle_W<<<nb((IN_DIM / 32) * 1024, B), B, 0, stream>>>(W1, IN_DIM, w1s);
    swizzle_W<<<nb((HID / 32) * 1024, B), B, 0, stream>>>(W2, HID, w2s);

    // ---- layer 1: WMMA GEMM (scaled by norm_out) -> SpMM -> relu epilogue
    gemm_nodes_bf16<IN_DIM><<<N_NODES / 16, 128, 0, stream>>>(xb, w1s, norm_out, t);
    zero_f32<<<nb((long long)N_NODES * HID, B), B, 0, stream>>>(m, N_NODES * HID);
    spmm_kernel<<<nb((long long)N_EDGES * 32, B), B, 0, stream>>>(ei, t, m);
    finish_layer_kernel<<<nb((long long)N_NODES * HID, B), B, 0, stream>>>(m, norm_in, b1, h1);

    // ---- layer 2
    gemm_nodes_bf16<HID><<<N_NODES / 16, 128, 0, stream>>>(h1, w2s, norm_out, t);
    zero_f32<<<nb((long long)N_NODES * HID, B), B, 0, stream>>>(m, N_NODES * HID);
    spmm_kernel<<<nb((long long)N_EDGES * 32, B), B, 0, stream>>>(ei, t, m);

    // ---- attention gate + per-graph pooling + classifier
    zero_f32<<<nb(N_GRAPHS * HID, B), B, 0, stream>>>(gsum, N_GRAPHS * HID);
    zero_f32<<<1, 64, 0, stream>>>(gcnt, N_GRAPHS);
    finish2_attn_pool_kernel<<<N_NODES / 8, 256, 0, stream>>>(
        m, norm_in, b2, attn_W, attn_b, gid, out_hw, gsum, gcnt);
    classify_kernel<<<1, 640, 0, stream>>>(gsum, gcnt, cls_W, cls_b, out_cls);
}
